// MultiHeadAttention_69630009803055
// MI455X (gfx1250) — matrix operand order repaired and checked by reading, not compiled, not run on hardware
//
#include <hip/hip_runtime.h>

// ---------------- types ----------------
typedef __attribute__((ext_vector_type(16))) __bf16 v16bf;
typedef __attribute__((ext_vector_type(8)))  __bf16 v8bf;
typedef __attribute__((ext_vector_type(8)))  float  v8f;

#define BB   16
#define TT   1024
#define DD   768
#define HH   12
#define HSZ  64
#define MTOT (BB*TT)      // 16384
#define NQKV (3*DD)       // 2304
#define KT32 (DD/32)      // 24 k-steps of 32

__device__ __forceinline__ __bf16 f2bf(float f) {
  unsigned u = __builtin_bit_cast(unsigned, f);
  unsigned r = (u + 0x7FFFu + ((u >> 16) & 1u)) >> 16;
  unsigned short h = (unsigned short)r;
  return __builtin_bit_cast(__bf16, h);
}

union FragA { v16bf v; struct { v8bf lo, hi; } p; };

#define WMMA_BF16(A, B, C) \
  __builtin_amdgcn_wmma_f32_16x16x32_bf16(false, (A), false, (B), (short)0, (C), false, false)

__device__ __forceinline__ void loadA(FragA& a, const __bf16* arow, int k0, int half) {
  a.p.lo = *(const v8bf*)(arow + k0 + half * 16);
  a.p.hi = *(const v8bf*)(arow + k0 + half * 16 + 8);
}
__device__ __forceinline__ void loadB4(v16bf b[4], const __bf16* bbase, int kt) {
#pragma unroll
  for (int nt = 0; nt < 4; ++nt)
    b[nt] = *(const v16bf*)(bbase + (size_t)(nt * KT32 + kt) * 512);
}

// 16x64 GEMM core with explicit double-buffered operands: loads for k-step t+1
// are in flight while WMMAs for k-step t execute (partial s_wait_loadcnt).
__device__ __forceinline__ void gemm_16x64(const __bf16* __restrict__ arow,
                                           const __bf16* __restrict__ bbase,
                                           int half, v8f c[4]) {
  FragA a0, a1;
  v16bf b0[4], b1[4];
  loadA(a0, arow, 0, half);
  loadB4(b0, bbase, 0);
  for (int kt = 0; kt < KT32 - 2; kt += 2) {
    loadA(a1, arow, (kt + 1) * 32, half);
    loadB4(b1, bbase, kt + 1);
#pragma unroll
    for (int nt = 0; nt < 4; ++nt) c[nt] = WMMA_BF16(a0.v, b0[nt], c[nt]);
    loadA(a0, arow, (kt + 2) * 32, half);
    loadB4(b0, bbase, kt + 2);
#pragma unroll
    for (int nt = 0; nt < 4; ++nt) c[nt] = WMMA_BF16(a1.v, b1[nt], c[nt]);
  }
  // epilogue: k-steps KT32-2, KT32-1
  loadA(a1, arow, (KT32 - 1) * 32, half);
  loadB4(b1, bbase, KT32 - 1);
#pragma unroll
  for (int nt = 0; nt < 4; ++nt) c[nt] = WMMA_BF16(a0.v, b0[nt], c[nt]);
#pragma unroll
  for (int nt = 0; nt < 4; ++nt) c[nt] = WMMA_BF16(a1.v, b1[nt], c[nt]);
}

// ---------------- fp32 -> bf16 convert (x) ----------------
__global__ __launch_bounds__(256) void cvt_x_kernel(const float* __restrict__ x,
                                                    __bf16* __restrict__ xb) {
  int i = blockIdx.x * 256 + threadIdx.x;          // one float4 each
  float4 v = ((const float4*)x)[i];
  __bf16* o = xb + (size_t)i * 4;
  o[0] = f2bf(v.x); o[1] = f2bf(v.y); o[2] = f2bf(v.z); o[3] = f2bf(v.w);
}

// ---------------- weight swizzle: per-lane-contiguous WMMA B fragments ----------------
// packed[((ntile*KT32 + kt)*32 + lane)*16 + e] = W[k = kt*32 + (lane>>4)*16 + e][n = ntile*16 + (lane&15)]
__global__ __launch_bounds__(256) void swz_qkv_kernel(const float* __restrict__ Wq,
                                                      const float* __restrict__ Wk,
                                                      const float* __restrict__ Wv,
                                                      __bf16* __restrict__ dst) {
  int idx = blockIdx.x * 256 + threadIdx.x;
  int e     = idx & 15;
  int lane  = (idx >> 4) & 31;
  int kt    = (idx >> 9) % KT32;
  int ntile = idx / (512 * KT32);
  int n = ntile * 16 + (lane & 15);
  int k = kt * 32 + (lane >> 4) * 16 + e;
  int which = n / DD;
  int nn = n - which * DD;
  int h = nn >> 6, ee = nn & 63;
  const float* W = (which == 0) ? Wq : (which == 1) ? Wk : Wv;   // (H, D, HS)
  dst[idx] = f2bf(W[((size_t)h * DD + k) * HSZ + ee]);
}

__global__ __launch_bounds__(256) void swz_wp_kernel(const float* __restrict__ Wp,
                                                     __bf16* __restrict__ dst) {
  int idx = blockIdx.x * 256 + threadIdx.x;
  int e     = idx & 15;
  int lane  = (idx >> 4) & 31;
  int kt    = (idx >> 9) % KT32;
  int ntile = idx / (512 * KT32);
  int n = ntile * 16 + (lane & 15);
  int k = kt * 32 + (lane >> 4) * 16 + e;
  dst[idx] = f2bf(Wp[(size_t)k * DD + n]);     // (D, D) row-major
}

// ---------------- fused QKV projection GEMM: one wave = 16x64 tile ----------------
// q,k -> [B][H][T][HS] bf16 ; v -> transposed [B][H][HS][T] bf16
__global__ __launch_bounds__(256)
void qkv_gemm_kernel(const __bf16* __restrict__ xb, const __bf16* __restrict__ wpk,
                     const float* __restrict__ bq, const float* __restrict__ bk,
                     const float* __restrict__ bv,
                     __bf16* __restrict__ qb, __bf16* __restrict__ kb,
                     __bf16* __restrict__ vtb) {
  const int wave = blockIdx.x * 8 + (threadIdx.x >> 5);
  const int lane = threadIdx.x & 31;
  const int half = lane >> 4, lm = lane & 15;
  const int ngroups = NQKV / 64;                  // 36 groups of 4 n-tiles
  const int mtile = wave / ngroups;
  const int ng = wave - mtile * ngroups;
  const int ntile0 = ng * 4;
  const int m0 = mtile * 16;

  const __bf16* arow  = xb  + (size_t)(m0 + lm) * DD;
  const __bf16* bbase = wpk + (size_t)ntile0 * KT32 * 512 + (size_t)lane * 16;

  v8f c[4] = {v8f{}, v8f{}, v8f{}, v8f{}};
  gemm_16x64(arow, bbase, half, c);

#pragma unroll
  for (int nt = 0; nt < 4; ++nt) {
    const int ntile = ntile0 + nt;
    const int which = ntile / 48;                 // group of 4 never straddles regions
    const int col = (ntile % 48) * 16 + lm;       // 0..767 within region
    const int h = col >> 6, e = col & 63;
    const float bias = (which == 0) ? bq[col] : (which == 1) ? bk[col] : bv[col];
#pragma unroll
    for (int r = 0; r < 8; ++r) {
      const int mg = m0 + half * 8 + r;           // global row in [0, B*T)
      const int b = mg >> 10, t = mg & (TT - 1);
      const __bf16 val = f2bf(c[nt][r] + bias);
      if (which == 0)
        qb[((size_t)(b * HH + h) * TT + t) * HSZ + e] = val;
      else if (which == 1)
        kb[((size_t)(b * HH + h) * TT + t) * HSZ + e] = val;
      else
        vtb[((size_t)(b * HH + h) * HSZ + e) * TT + t] = val;
    }
  }
}

// ---------------- flash attention: one wave = 16 query rows, Bc = 64 ----------------
__global__ __launch_bounds__(256)
void flash_attn_kernel(const __bf16* __restrict__ qb, const __bf16* __restrict__ kb,
                       const __bf16* __restrict__ vtb, __bf16* __restrict__ attnb) {
  __shared__ __bf16 plds[8 * 16 * 64];            // 2KB per wave
  const int wid = threadIdx.x >> 5;
  const int wave = blockIdx.x * 8 + wid;
  const int lane = threadIdx.x & 31;
  const int half = lane >> 4, lm = lane & 15;
  const int ttile = wave & 63;
  const int bh = wave >> 6;                       // 0..B*H-1
  const int t0 = ttile * 16;

  const __bf16* qp = qb  + (size_t)bh * TT * HSZ;
  const __bf16* kp = kb  + (size_t)bh * TT * HSZ;
  const __bf16* vp = vtb + (size_t)bh * HSZ * TT;

  FragA qa0, qa1;
  {
    const __bf16* qrow = qp + (size_t)(t0 + lm) * HSZ;
    qa0.p.lo = *(const v8bf*)(qrow + half * 16);
    qa0.p.hi = *(const v8bf*)(qrow + 8 + half * 16);
    qa1.p.lo = *(const v8bf*)(qrow + 32 + half * 16);
    qa1.p.hi = *(const v8bf*)(qrow + 40 + half * 16);
  }

  v8f o[4] = {v8f{}, v8f{}, v8f{}, v8f{}};
  float mrow[8], lrow[8];
#pragma unroll
  for (int r = 0; r < 8; ++r) { mrow[r] = -3.0e38f; lrow[r] = 0.0f; }

  __bf16* myp = plds + wid * (16 * 64);
  const float scale = 0.125f;                     // 1/sqrt(64)

  for (int s0 = 0; s0 < TT; s0 += 64) {
    // ---- load all 8 K fragments, then 8 S-WMMAs (partial loadcnt waits)
    v16bf kf[8];
#pragma unroll
    for (int st = 0; st < 4; ++st) {
      const __bf16* kn = kp + (size_t)(s0 + st * 16 + lm) * HSZ;
      kf[2 * st]     = *(const v16bf*)(kn + half * 16);
      kf[2 * st + 1] = *(const v16bf*)(kn + 32 + half * 16);
    }
    v8f s[4];
#pragma unroll
    for (int st = 0; st < 4; ++st) {
      v8f acc = {};
      acc = WMMA_BF16(qa0.v, kf[2 * st], acc);
      acc = WMMA_BF16(qa1.v, kf[2 * st + 1], acc);
      s[st] = acc;
    }
    // ---- issue V loads now; softmax math below hides their latency
    v16bf vf[8];
#pragma unroll
    for (int nt = 0; nt < 4; ++nt) {
      const __bf16* vn = vp + (size_t)(nt * 16 + lm) * TT + s0 + half * 16;
      vf[2 * nt]     = *(const v16bf*)vn;
      vf[2 * nt + 1] = *(const v16bf*)(vn + 32);
    }
    // ---- online softmax over 64 columns (rows live across lane groups of 16)
#pragma unroll
    for (int r = 0; r < 8; ++r) {
      float sv[4];
      float mx = -3.0e38f;
#pragma unroll
      for (int st = 0; st < 4; ++st) { sv[st] = s[st][r] * scale; mx = fmaxf(mx, sv[st]); }
#pragma unroll
      for (int msk = 1; msk < 16; msk <<= 1) mx = fmaxf(mx, __shfl_xor(mx, msk, 32));
      float mnew = fmaxf(mrow[r], mx);
      float alpha = __expf(mrow[r] - mnew);
      float rs = 0.0f;
#pragma unroll
      for (int st = 0; st < 4; ++st) {
        float p = __expf(sv[st] - mnew);
        rs += p;
        myp[(half * 8 + r) * 64 + st * 16 + lm] = f2bf(p);   // C-layout -> row-major LDS
      }
#pragma unroll
      for (int msk = 1; msk < 16; msk <<= 1) rs += __shfl_xor(rs, msk, 32);
      lrow[r] = lrow[r] * alpha + rs;
      mrow[r] = mnew;
#pragma unroll
      for (int nt = 0; nt < 4; ++nt) o[nt][r] *= alpha;
    }
    asm volatile("s_wait_dscnt 0" ::: "memory");
    FragA pa0, pa1;                               // A-fragments of P: 16 x 64 over s
    pa0.p.lo = *(const v8bf*)(myp + lm * 64 + half * 16);
    pa0.p.hi = *(const v8bf*)(myp + lm * 64 + 8 + half * 16);
    pa1.p.lo = *(const v8bf*)(myp + lm * 64 + 32 + half * 16);
    pa1.p.hi = *(const v8bf*)(myp + lm * 64 + 40 + half * 16);
    // ---- O += P * V   (V fragments already resident)
#pragma unroll
    for (int nt = 0; nt < 4; ++nt) {
      o[nt] = WMMA_BF16(pa0.v, vf[2 * nt], o[nt]);
      o[nt] = WMMA_BF16(pa1.v, vf[2 * nt + 1], o[nt]);
    }
  }

  // ---- normalize + store to [B*T][D] bf16 (head-concat layout)
  const int b = bh / HH, h = bh - b * HH;
#pragma unroll
  for (int r = 0; r < 8; ++r) {
    float inv = 1.0f / lrow[r];
    const int t = t0 + half * 8 + r;
    __bf16* orow = attnb + (size_t)(b * TT + t) * DD + h * HSZ;
#pragma unroll
    for (int nt = 0; nt < 4; ++nt)
      orow[nt * 16 + lm] = f2bf(o[nt][r] * inv);
  }
}

// ---------------- output projection GEMM: one wave = 16x64 tile, fp32 out ----------------
__global__ __launch_bounds__(256)
void proj_gemm_kernel(const __bf16* __restrict__ attnb, const __bf16* __restrict__ wpp,
                      const float* __restrict__ bp, float* __restrict__ out) {
  const int wave = blockIdx.x * 8 + (threadIdx.x >> 5);
  const int lane = threadIdx.x & 31;
  const int half = lane >> 4, lm = lane & 15;
  const int ngroups = DD / 64;                    // 12 groups of 4 n-tiles
  const int mtile = wave / ngroups;
  const int ng = wave - mtile * ngroups;
  const int ntile0 = ng * 4;
  const int m0 = mtile * 16;

  const __bf16* arow  = attnb + (size_t)(m0 + lm) * DD;
  const __bf16* bbase = wpp + (size_t)ntile0 * KT32 * 512 + (size_t)lane * 16;

  v8f c[4] = {v8f{}, v8f{}, v8f{}, v8f{}};
  gemm_16x64(arow, bbase, half, c);

#pragma unroll
  for (int nt = 0; nt < 4; ++nt) {
    const float bias = bp[(ntile0 + nt) * 16 + lm];
#pragma unroll
    for (int r = 0; r < 8; ++r) {
      const int mg = m0 + half * 8 + r;
      out[(size_t)mg * DD + (ntile0 + nt) * 16 + lm] = c[nt][r] + bias;
    }
  }
}

// ---------------- host launch ----------------
extern "C" void kernel_launch(void* const* d_in, const int* in_sizes, int n_in,
                              void* d_out, int out_size, void* d_ws, size_t ws_size,
                              hipStream_t stream) {
  const float* x  = (const float*)d_in[0];
  const float* Wq = (const float*)d_in[1];
  const float* bq = (const float*)d_in[2];
  const float* Wk = (const float*)d_in[3];
  const float* bk = (const float*)d_in[4];
  const float* Wv = (const float*)d_in[5];
  const float* bv = (const float*)d_in[6];
  const float* Wp = (const float*)d_in[7];
  const float* bp = (const float*)d_in[8];
  float* out = (float*)d_out;

  char* base = (char*)d_ws;
  __bf16* xb   = (__bf16*)base; base += (size_t)MTOT * DD * 2;
  __bf16* wqk  = (__bf16*)base; base += (size_t)NQKV * DD * 2;
  __bf16* wpp  = (__bf16*)base; base += (size_t)DD * DD * 2;
  __bf16* qb   = (__bf16*)base; base += (size_t)MTOT * DD * 2;
  __bf16* kb   = (__bf16*)base; base += (size_t)MTOT * DD * 2;
  __bf16* vtb  = (__bf16*)base; base += (size_t)MTOT * DD * 2;
  __bf16* attn = (__bf16*)base;

  cvt_x_kernel<<<(MTOT * DD / 4) / 256, 256, 0, stream>>>(x, xb);
  swz_qkv_kernel<<<(NQKV * DD) / 256, 256, 0, stream>>>(Wq, Wk, Wv, wqk);
  swz_wp_kernel<<<(DD * DD) / 256, 256, 0, stream>>>(Wp, wpp);
  qkv_gemm_kernel<<<(MTOT / 16) * (NQKV / 64) / 8, 256, 0, stream>>>(
      xb, wqk, bq, bk, bv, qb, kb, vtb);
  flash_attn_kernel<<<(BB * HH * (TT / 16)) / 8, 256, 0, stream>>>(qb, kb, vtb, attn);
  proj_gemm_kernel<<<(MTOT / 16) * (DD / 64) / 8, 256, 0, stream>>>(attn, wpp, bp, out);
}
